// RelativeMultiHeadAttention_40793599377983
// MI455X (gfx1250) — compile-verified, matrix operand order checked by reading, not run on hardware
//
#include <hip/hip_runtime.h>
#include <hip/hip_bf16.h>

// ---------------------------------------------------------------------------
// Transformer-XL relative multi-head attention for MI455X (gfx1250, wave32).
// All GEMMs via v_wmma_f32_16x16x32_bf16 (fp32 accumulate).
// B=2, Q=1024, MEM=1024, K=2048, D=1024, H=16, dk=64.
// ---------------------------------------------------------------------------

namespace {
constexpr int BATCH = 2;
constexpr int QL    = 1024;   // query length
constexpr int ML    = 1024;   // memory length
constexpr int KL    = 2048;   // key length = QL + ML
constexpr int KLP   = 2080;   // padded k-stride for transposed V (16B aligned)
constexpr int DM    = 1024;   // model dim
constexpr int NH    = 16;     // heads
constexpr int DK    = 64;     // head dim
}

typedef __attribute__((ext_vector_type(16))) __bf16 v16bf;
typedef __attribute__((ext_vector_type(8)))  __bf16 v8bf;
typedef __attribute__((ext_vector_type(4)))  __bf16 v4bf;
typedef __attribute__((ext_vector_type(8)))  float  v8f;

// ISA 16-bit A/B fragment element -> K-index mapping (CDNA5 ISA):
// lanes 0-15: elems 0..7 -> K 0..7,  elems 8..15 -> K 16..23
// lanes16-31: elems 0..7 -> K 8..15, elems 8..15 -> K 24..31
__device__ __forceinline__ v8f wmma_bf16(v16bf a, v16bf b, v8f c) {
  return __builtin_amdgcn_wmma_f32_16x16x32_bf16(
      false, a, false, b, (short)0, c, false, false);
}

// Load a 16-element fragment whose 32-wide K-slice is contiguous at `base`.
// Two 16-byte vector loads (kk = half*8 .. and kk = 16+half*8 ..).
__device__ __forceinline__ v16bf load_frag(const __bf16* base, int half) {
  const v8bf lo = *(const v8bf*)(base + half * 8);
  const v8bf hi = *(const v8bf*)(base + 16 + half * 8);
  v16bf f;
  #pragma unroll
  for (int e = 0; e < 8; ++e) { f[e] = lo[e]; f[8 + e] = hi[e]; }
  return f;
}

// ---------------------------------------------------------------------------
// 32x64 macro-tile GEMM, 8 waves (2x4), one 16x16 WMMA tile per wave.
// MODE 0: concat(mem,inputs) @ W -> qu/qv (u,v folded), K, V^T (bf16)
// MODE 1: r @ W_r               -> rb (bf16)
// MODE 2: attn(bf16) @ W_o      -> out (fp32)
// ---------------------------------------------------------------------------
template<int MODE>
__global__ __launch_bounds__(256) void gemm_tile(
    const float* __restrict__ A0, const float* __restrict__ A1,
    const __bf16* __restrict__ Abf,
    const float* __restrict__ Bmat,
    int Kdim, int Ncols,
    const float* __restrict__ u, const float* __restrict__ v,
    __bf16* __restrict__ quw, __bf16* __restrict__ qvw,
    __bf16* __restrict__ kbw, __bf16* __restrict__ vbT,
    __bf16* __restrict__ rbw, float* __restrict__ outw)
{
  __shared__ __align__(16) __bf16 ldsA[32][48];    // (row, k), 96B stride
  __shared__ __align__(16) __bf16 ldsBT[64][48];   // (col, k), transposed

  const int tid  = threadIdx.x;
  const int wv   = tid >> 5;
  const int lane = tid & 31;
  const int half = lane >> 4;
  const int idx  = lane & 15;
  const int rw   = wv >> 2;          // wave row 0..1
  const int cw   = wv & 3;           // wave col 0..3

  const int b  = blockIdx.z;
  const int m0 = blockIdx.y * 32;
  const int n0 = blockIdx.x * 64;

  v8f acc = (v8f)(0.0f);

  for (int k0 = 0; k0 < Kdim; k0 += 32) {
    // --- stage A (32x32): one float4 / v4bf per thread, coalesced ---
    {
      int row = tid >> 3, col4 = tid & 7;
      int rg  = m0 + row;
      if (MODE == 2) {
        v4bf a = *(const v4bf*)(Abf + (size_t)rg * Kdim + k0 + col4 * 4);
        *(v4bf*)(&ldsA[row][col4 * 4]) = a;
      } else {
        const float* src;
        if (MODE == 0) {
          src = (rg < ML) ? (A0 + ((size_t)b * ML + rg) * DM)
                          : (A1 + ((size_t)b * QL + (rg - ML)) * DM);
        } else {
          src = A0 + (size_t)rg * Kdim;
        }
        float4 f4 = *(const float4*)(src + k0 + col4 * 4);
        v4bf p;
        p[0] = (__bf16)f4.x; p[1] = (__bf16)f4.y;
        p[2] = (__bf16)f4.z; p[3] = (__bf16)f4.w;
        *(v4bf*)(&ldsA[row][col4 * 4]) = p;
      }
    }
    // --- stage B (32x64) transposed: two float4 per thread ---
    #pragma unroll
    for (int i = 0; i < 2; ++i) {
      int e = tid + 256 * i;
      int row = e >> 4, col4 = e & 15;
      float4 f4 = *(const float4*)(Bmat + (size_t)(k0 + row) * Ncols + n0 + col4 * 4);
      ldsBT[col4 * 4 + 0][row] = (__bf16)f4.x;
      ldsBT[col4 * 4 + 1][row] = (__bf16)f4.y;
      ldsBT[col4 * 4 + 2][row] = (__bf16)f4.z;
      ldsBT[col4 * 4 + 3][row] = (__bf16)f4.w;
    }
    __syncthreads();

    v16bf aF = load_frag(&ldsA[rw * 16 + idx][0], half);
    v16bf bF = load_frag(&ldsBT[cw * 16 + idx][0], half);
    acc = wmma_bf16(aF, bF, acc);
    __syncthreads();
  }

  // --- write C tile: reg g <-> row g+8*half, lane <-> col idx ---
  const int orow0 = m0 + rw * 16;
  const int ocol  = n0 + cw * 16 + idx;
  #pragma unroll
  for (int g = 0; g < 8; ++g) {
    int orow  = orow0 + g + 8 * half;
    float val = acc[g];
    if (MODE == 0) {
      int s = ocol >> 10, cc = ocol & 1023;
      if (s == 0) {                 // q-heads: keep only rows >= ML
        if (orow >= ML) {
          size_t off = ((size_t)b * QL + (orow - ML)) * DM + cc;
          quw[off] = (__bf16)(val + u[cc]);
          qvw[off] = (__bf16)(val + v[cc]);
        }
      } else if (s == 1) {
        kbw[((size_t)b * KL + orow) * DM + cc] = (__bf16)val;
      } else {                      // V stored transposed: (b, h, d, k)
        int hh = cc >> 6, dd = cc & 63;
        vbT[((size_t)(b * NH + hh) * DK + dd) * KLP + orow] = (__bf16)val;
      }
    } else if (MODE == 1) {
      rbw[(size_t)orow * DM + ocol] = (__bf16)val;
    } else {
      outw[(size_t)orow * DM + ocol] = val;
    }
  }
}

// ---------------------------------------------------------------------------
// Fused flash attention with analytic rel_shift.
// One wave per 16-row q-tile; 8 independent waves per block.
// rel_shift(bd)[i,j] = (q_i+v) . r_proj[j-i+ML-1]   (valid/causal region)
// Per 16x16 k-tile the relative index spans a 31-wide band: compute a 16x32
// BD band with WMMA, spill to per-wave LDS, gather the diagonal.
// ---------------------------------------------------------------------------
__global__ __launch_bounds__(256) void attn_fused(
    const __bf16* __restrict__ qu, const __bf16* __restrict__ qv,
    const __bf16* __restrict__ kb, const __bf16* __restrict__ vbT,
    const __bf16* __restrict__ rb, __bf16* __restrict__ attn)
{
  __shared__ __align__(16) float  bdS[8][16][33];   // per-wave BD band
  __shared__ __align__(16) __bf16 pS[8][16][16];    // per-wave P transpose

  const int wv   = threadIdx.x >> 5;
  const int lane = threadIdx.x & 31;
  const int half = lane >> 4;
  const int idx  = lane & 15;

  const int bh = blockIdx.y;          // b*NH + h
  const int b  = bh >> 4;
  const int h  = bh & 15;
  const int i0 = (blockIdx.x * 8 + wv) * 16;

  // --- Q fragments (A-layout), dk=64 -> 2 k-steps of 32, loaded once ---
  v16bf aqu[2], aqv[2];
  {
    const __bf16* qbase = qu + ((size_t)(b * QL + i0 + idx)) * DM + h * DK;
    const __bf16* vbase = qv + ((size_t)(b * QL + i0 + idx)) * DM + h * DK;
    #pragma unroll
    for (int ks = 0; ks < 2; ++ks) {
      aqu[ks] = load_frag(qbase + ks * 32, half);
      aqv[ks] = load_frag(vbase + ks * 32, half);
    }
  }

  float mrow[8], lrow[8], corr[8];
  v8f o[4];
  #pragma unroll
  for (int g = 0; g < 8; ++g) { mrow[g] = -1e30f; lrow[g] = 0.0f; }
  #pragma unroll
  for (int t = 0; t < 4; ++t) o[t] = (v8f)(0.0f);

  const __bf16* vhead = vbT + (size_t)(b * NH + h) * DK * KLP;

  const int ntiles = (i0 + 15 + ML) / 16 + 1;   // causal: j <= i + ML
  for (int jt = 0; jt < ntiles; ++jt) {
    const int j0    = jt * 16;
    const int mbase = j0 - i0 + (ML - 16);      // relative-index band start

    // prefetch K rows ~8 tiles ahead (global_prefetch_b8, speculative)
    if (j0 + 128 + idx < KL)
      __builtin_prefetch(kb + ((size_t)(b * KL + j0 + 128 + idx)) * DM + h * DK, 0, 1);

    // --- BD band: (q+v) @ r^T for m in [mbase, mbase+32) ---
    v8f bd0 = (v8f)(0.0f), bd1 = (v8f)(0.0f);
    {
      int m0r = mbase + idx;       m0r = m0r < 0 ? 0 : (m0r > KL - 1 ? KL - 1 : m0r);
      int m1r = mbase + 16 + idx;  m1r = m1r < 0 ? 0 : (m1r > KL - 1 ? KL - 1 : m1r);
      const __bf16* p0 = rb + (size_t)m0r * DM + h * DK;
      const __bf16* p1 = rb + (size_t)m1r * DM + h * DK;
      #pragma unroll
      for (int ks = 0; ks < 2; ++ks) {
        v16bf b0 = load_frag(p0 + ks * 32, half);
        v16bf b1 = load_frag(p1 + ks * 32, half);
        bd0 = wmma_bf16(aqv[ks], b0, bd0);
        bd1 = wmma_bf16(aqv[ks], b1, bd1);
      }
    }
    #pragma unroll
    for (int g = 0; g < 8; ++g) {
      bdS[wv][g + 8 * half][idx]      = bd0[g];
      bdS[wv][g + 8 * half][16 + idx] = bd1[g];
    }

    // --- AC: (q+u) @ k^T ---
    v8f s = (v8f)(0.0f);
    {
      const __bf16* kbase = kb + ((size_t)(b * KL + j0 + idx)) * DM + h * DK;
      #pragma unroll
      for (int ks = 0; ks < 2; ++ks)
        s = wmma_bf16(aqu[ks], load_frag(kbase + ks * 32, half), s);
    }

    // --- scores + diagonal BD gather + mask + online softmax ---
    #pragma unroll
    for (int g = 0; g < 8; ++g) {
      const int di = g + 8 * half;
      const int ig = i0 + di;
      const int jg = j0 + idx;
      float val = (s[g] + bdS[wv][di][idx - di + 15]) * 0.125f;  // 1/sqrt(64)
      if (jg > ig + ML) val = -1e30f;

      float rmax = val;
      rmax = fmaxf(rmax, __shfl_xor(rmax, 1, 32));
      rmax = fmaxf(rmax, __shfl_xor(rmax, 2, 32));
      rmax = fmaxf(rmax, __shfl_xor(rmax, 4, 32));
      rmax = fmaxf(rmax, __shfl_xor(rmax, 8, 32));

      float mnew = fmaxf(mrow[g], rmax);
      float c    = __expf(mrow[g] - mnew);
      float pe   = __expf(val - mnew);

      float rsum = pe;
      rsum += __shfl_xor(rsum, 1, 32);
      rsum += __shfl_xor(rsum, 2, 32);
      rsum += __shfl_xor(rsum, 4, 32);
      rsum += __shfl_xor(rsum, 8, 32);

      lrow[g] = lrow[g] * c + rsum;
      mrow[g] = mnew;
      corr[g] = c;
      pS[wv][di][idx] = (__bf16)pe;   // C-layout -> LDS for transpose
    }
    #pragma unroll
    for (int t = 0; t < 4; ++t)
      #pragma unroll
      for (int g = 0; g < 8; ++g) o[t][g] *= corr[g];

    // --- P @ V: A = P (16x16, zero-padded to K=32), B = V^T slices ---
    v16bf aP;
    {
      const v8bf plo = *(const v8bf*)(&pS[wv][idx][half * 8]);
      #pragma unroll
      for (int e = 0; e < 8; ++e) { aP[e] = plo[e]; aP[8 + e] = (__bf16)0.0f; }
    }
    #pragma unroll
    for (int t = 0; t < 4; ++t) {
      // contiguous along k thanks to transposed V; tail overhang (j0+16..31)
      // lands in the padded stride region and multiplies the zero half of aP
      v16bf bV = load_frag(vhead + (size_t)(t * 16 + idx) * KLP + j0, half);
      o[t] = wmma_bf16(aP, bV, o[t]);
    }
  }

  // --- normalize and write attn (bf16) ---
  #pragma unroll
  for (int t = 0; t < 4; ++t)
    #pragma unroll
    for (int g = 0; g < 8; ++g) {
      int di = g + 8 * half;
      attn[((size_t)(b * QL + i0 + di)) * DM + h * DK + t * 16 + idx] =
          (__bf16)(o[t][g] / lrow[g]);
    }
}

// ---------------------------------------------------------------------------
extern "C" void kernel_launch(void* const* d_in, const int* in_sizes, int n_in,
                              void* d_out, int out_size, void* d_ws, size_t ws_size,
                              hipStream_t stream) {
  (void)in_sizes; (void)n_in; (void)out_size; (void)ws_size;
  const float* inputs = (const float*)d_in[0];
  const float* mem    = (const float*)d_in[1];
  const float* r      = (const float*)d_in[2];
  const float* W      = (const float*)d_in[3];
  const float* W_r    = (const float*)d_in[4];
  const float* W_o    = (const float*)d_in[5];
  const float* u      = (const float*)d_in[6];
  const float* v      = (const float*)d_in[7];
  float* out = (float*)d_out;

  char* ws = (char*)d_ws;                         // ~33 MB bf16 intermediates
  __bf16* quw   = (__bf16*)(ws +  (0ull << 20));  // (B,Q,D)   q+u
  __bf16* qvw   = (__bf16*)(ws +  (4ull << 20));  // (B,Q,D)   q+v
  __bf16* kbw   = (__bf16*)(ws +  (8ull << 20));  // (B,K,D)
  __bf16* vbT   = (__bf16*)(ws + (16ull << 20));  // (B,H,dk,KLP) V transposed
  __bf16* rbw   = (__bf16*)(ws + (25ull << 20));  // (K,D)
  __bf16* attnw = (__bf16*)(ws + (29ull << 20));  // (B,Q,D)

  dim3 blk(256);

  // 1) fused QKV projection from concat(mem, inputs); u/v folded into q
  gemm_tile<0><<<dim3(3 * DM / 64, KL / 32, BATCH), blk, 0, stream>>>(
      mem, inputs, nullptr, W, DM, 3 * DM,
      u, v, quw, qvw, kbw, vbT, nullptr, nullptr);

  // 2) relative-position projection r @ W_r
  gemm_tile<1><<<dim3(DM / 64, KL / 32, 1), blk, 0, stream>>>(
      r, nullptr, nullptr, W_r, DM, DM,
      nullptr, nullptr, nullptr, nullptr, nullptr, nullptr, rbw, nullptr);

  // 3) fused flash attention with analytic rel_shift
  attn_fused<<<dim3(QL / 128, BATCH * NH, 1), blk, 0, stream>>>(
      quw, qvw, kbw, vbT, rbw, attnw);

  // 4) output projection attn @ W_o -> fp32
  gemm_tile<2><<<dim3(DM / 64, (BATCH * QL) / 32, 1), blk, 0, stream>>>(
      nullptr, nullptr, attnw, W_o, DM, DM,
      nullptr, nullptr, nullptr, nullptr, nullptr, nullptr, nullptr, out);
}
